// VectorQuantizer_90082644066445
// MI455X (gfx1250) — compile-verified, matrix operand order checked by reading, not run on hardware
//
#include <hip/hip_runtime.h>
#include <hip/hip_bf16.h>

typedef __attribute__((ext_vector_type(16))) _Float16 v16h;
typedef __attribute__((ext_vector_type(8)))  _Float16 v8h;
typedef __attribute__((ext_vector_type(8)))  float    v8f;
typedef __attribute__((ext_vector_type(4)))  int      v4i;

#define AS1 __attribute__((address_space(1)))
#define AS3 __attribute__((address_space(3)))

#define BB    64
#define DD    128
#define TT    1024
#define KK    1024
#define NN    (BB*TT)      // 65536 rows
#define KC2   256          // 2-term compensated contraction = 2*D
#define NSTEP 8            // KC2 / 32 WMMA k-steps
#define MT    128          // rows per workgroup
#define RT    2            // row-tiles per wave (32 rows/wave, 4 waves)
#define ASTR  264          // padded LDS row stride in f16 elems (528B, 16B-aligned)
#define NBLK  16           // codes per iteration
#define EPSF  1e-8f

#if __has_builtin(__builtin_amdgcn_global_load_async_to_lds_b128)
#define HAS_ASYNC_LDS 1
#else
#define HAS_ASYNC_LDS 0
#endif

// 16-byte global -> LDS copy; async (ASYNCcnt-tracked) when available.
__device__ __forceinline__ void copy16_g2l(const void* g, void* l) {
#if HAS_ASYNC_LDS
  __builtin_amdgcn_global_load_async_to_lds_b128((AS1 v4i*)g, (AS3 v4i*)l, 0, 0);
#else
  *(uint4*)l = *(const uint4*)g;
#endif
}

__device__ __forceinline__ void wait_async_then_barrier() {
#if HAS_ASYNC_LDS
#if __has_builtin(__builtin_amdgcn_s_wait_asynccnt)
  __builtin_amdgcn_s_wait_asynccnt(0);
#else
  asm volatile("s_wait_asynccnt 0x0" ::: "memory");
#endif
#endif
  __syncthreads();
}

// ---------------------------------------------------------------------------
// prep: codebook -> negated f16 B' = [-c_hi | -c_hi] [K][256],
// cnorm = 0.5*||c_hi||^2 (consistent with the rounded operand),
// zero the histogram and SSE accumulator.
// ---------------------------------------------------------------------------
__global__ void vq_prep(const float* __restrict__ cb, _Float16* __restrict__ Bc,
                        float* __restrict__ cnorm, float* __restrict__ counts,
                        float* __restrict__ sse) {
  const int k = blockIdx.x;
  const int d = threadIdx.x;              // 128 threads = 128 dims
  float v = cb[k * DD + d];
  _Float16 hi = (_Float16)v;
  float    hf = (float)hi;
  _Float16 nh = (_Float16)(-hf);
  Bc[k * KC2 +       d] = nh;             // pairs with z_hi
  Bc[k * KC2 + 128 + d] = nh;             // pairs with z_lo

  __shared__ float red[128];
  red[d] = 0.5f * hf * hf;
  __syncthreads();
  for (int s = 64; s > 0; s >>= 1) {
    if (d < s) red[d] += red[d + s];
    __syncthreads();
  }
  if (d == 0) {
    cnorm[k]  = red[0];
    counts[k] = 0.0f;
    if (k == 0) *sse = 0.0f;
  }
}

// ---------------------------------------------------------------------------
// main: 512 blocks x 128 threads (4 waves). Each wave owns 2 row-tiles
// (32 rows) resident in VGPRs; every B fragment feeds 2 WMMAs (halves DS/WMMA).
// 64 code-blocks of 16; 8 f16 WMMA k-steps per tile per block.
// LDS B-tile double-buffered via async global->LDS; B frags reg-double-buffered.
// ---------------------------------------------------------------------------
__global__ __launch_bounds__(128) void vq_main(
    const float* __restrict__ ze, const float* __restrict__ cb,
    const _Float16* __restrict__ Bc, const float* __restrict__ cnorm,
    float* __restrict__ counts, float* __restrict__ sseg,
    float* __restrict__ zq_out, float* __restrict__ idx_out) {
  __shared__ __align__(16) _Float16 Alds[MT * ASTR];        // ~66 KB
  __shared__ __align__(16) _Float16 Blds[2][NBLK * ASTR];   // 2 x ~8.3 KB
  __shared__ int   idx_lds[MT];
  __shared__ float red4[4];

  const int tid = threadIdx.x;
  const int n0  = blockIdx.x * MT;
  const int b   = n0 >> 10;       // T = 1024
  const int t0  = n0 & 1023;

  // ---- stage A' = [z_hi | z_lo] into LDS; float4 along t (coalesced) ----
  {
    const int t4 = (tid & 31) * 4;            // 4 consecutive t per thread
    for (int d = tid >> 5; d < DD; d += 4) {  // 4 d-rows per pass
      float4 v = *(const float4*)&ze[((b * DD + d) << 10) + t0 + t4];
      float vv[4] = {v.x, v.y, v.z, v.w};
#pragma unroll
      for (int j = 0; j < 4; ++j) {
        _Float16 hi = (_Float16)vv[j];
        _Float16 lo = (_Float16)(vv[j] - (float)hi);
        Alds[(t4 + j) * ASTR +       d] = hi;
        Alds[(t4 + j) * ASTR + 128 + d] = lo;
      }
    }
  }

  // ---- stage B'-tile for kb=0 into buffer 0 (overlaps with A staging) ----
  {
    const uint4* src = (const uint4*)Bc;
#pragma unroll
    for (int j = 0; j < 4; ++j) {             // 512 16B chunks / 128 threads
      int i = tid + j * 128;
      int row = i / (KC2 / 8);
      int col = i % (KC2 / 8);
      copy16_g2l(src + i, &Blds[0][row * ASTR + col * 8]);
    }
  }
  wait_async_then_barrier();

  const int lane = tid & 31;
  const int wave = tid >> 5;    // 0..3, owns rows [wave*32, wave*32+32)
  const int half = lane >> 4;
  const int l16  = lane & 15;

  // ---- load A fragments for both row-tiles, keep resident (ISA 16-bit A
  //      16x32 layout: per-lane chunks at K = half*8 and K = 16+half*8) ----
  union Frag { v16h v; v8h h[2]; };
  Frag afrag[RT][NSTEP];
#pragma unroll
  for (int p = 0; p < RT; ++p) {
    const _Float16* arow = &Alds[(wave * 32 + p * 16 + l16) * ASTR];
#pragma unroll
    for (int s = 0; s < NSTEP; ++s) {
      afrag[p][s].h[0] = *(const v8h*)(arow + s * 32 +      half * 8);
      afrag[p][s].h[1] = *(const v8h*)(arow + s * 32 + 16 + half * 8);
    }
  }

  float minv[RT][8];
  int   mini[RT][8];
#pragma unroll
  for (int p = 0; p < RT; ++p)
#pragma unroll
    for (int i = 0; i < 8; ++i) { minv[p][i] = 3.4e38f; mini[p][i] = 0; }

  for (int kb = 0; kb < KK / NBLK; ++kb) {
    const int cur = kb & 1;

    // ---- stage next B'-tile into the other buffer (hidden behind WMMAs) ----
    if (kb + 1 < KK / NBLK) {
      const uint4* src = (const uint4*)(Bc + (size_t)(kb + 1) * NBLK * KC2);
#pragma unroll
      for (int j = 0; j < 4; ++j) {
        int i = tid + j * 128;
        int row = i / (KC2 / 8);
        int col = i % (KC2 / 8);
        copy16_g2l(src + i, &Blds[cur ^ 1][row * ASTR + col * 8]);
      }
    }
    // warm L2 for block kb+2 (64 cachelines of 128B)
    if (kb + 2 < KK / NBLK && tid < 64) {
      __builtin_prefetch((const char*)Bc + ((size_t)(kb + 2) * NBLK * KC2) * 2 +
                             (size_t)tid * 128, 0, 0);
    }

    const int   code = kb * NBLK + l16;
    const float cn   = cnorm[code];

    // ---- 8-step WMMA chains: each B fragment feeds both row-tiles ----
    v8f acc0 = {}, acc1 = {};
    const _Float16* brow = &Blds[cur][l16 * ASTR];  // lane's code column
    Frag bfr[2];
    bfr[0].h[0] = *(const v8h*)(brow +      half * 8);
    bfr[0].h[1] = *(const v8h*)(brow + 16 + half * 8);
#pragma unroll
    for (int s = 0; s < NSTEP; ++s) {
      if (s < NSTEP - 1) {
        bfr[(s + 1) & 1].h[0] = *(const v8h*)(brow + (s + 1) * 32 +      half * 8);
        bfr[(s + 1) & 1].h[1] = *(const v8h*)(brow + (s + 1) * 32 + 16 + half * 8);
      }
      acc0 = __builtin_amdgcn_wmma_f32_16x16x32_f16(
          false, afrag[0][s].v, false, bfr[s & 1].v, (short)0, acc0, false, false);
      acc1 = __builtin_amdgcn_wmma_f32_16x16x32_f16(
          false, afrag[1][s].v, false, bfr[s & 1].v, (short)0, acc1, false, false);
    }

#pragma unroll
    for (int i = 0; i < 8; ++i) {
      float s0 = acc0[i] + cn;              // 0.5||c_hi||^2 - z.c_hi
      bool  m0 = s0 < minv[0][i];
      minv[0][i] = m0 ? s0 : minv[0][i];
      mini[0][i] = m0 ? code : mini[0][i];
      float s1 = acc1[i] + cn;
      bool  m1 = s1 < minv[1][i];
      minv[1][i] = m1 ? s1 : minv[1][i];
      mini[1][i] = m1 ? code : mini[1][i];
    }

    // async stores for kb+1 done + everyone finished reading buffer `cur`
    wait_async_then_barrier();
  }

  // ---- cross-lane argmin over the 16 lanes of each half ----
#pragma unroll
  for (int p = 0; p < RT; ++p) {
#pragma unroll
    for (int m = 8; m >= 1; m >>= 1) {
#pragma unroll
      for (int i = 0; i < 8; ++i) {
        float ov = __shfl_xor(minv[p][i], m, 32);
        int   oi = __shfl_xor(mini[p][i], m, 32);
        bool  take = (ov < minv[p][i]) || (ov == minv[p][i] && oi < mini[p][i]);
        minv[p][i] = take ? ov : minv[p][i];
        mini[p][i] = take ? oi : mini[p][i];
      }
    }
  }
  if (l16 == 0) {
#pragma unroll
    for (int p = 0; p < RT; ++p) {
#pragma unroll
      for (int i = 0; i < 8; ++i) {
        int row = wave * 32 + p * 16 + half * 8 + i;  // C/D layout rows
        idx_lds[row] = mini[p][i];
        idx_out[n0 + row] = (float)mini[p][i];
        atomicAdd(&counts[mini[p][i]], 1.0f);
      }
    }
  }
  __syncthreads();

  // ---- z_q gather/write + SSE (coalesced along t) ----
  float sse = 0.0f;
  {
    const int r = tid;                      // 128 threads = 128 rows
    const int code = idx_lds[r];
    for (int d = 0; d < DD; ++d) {
      float q = cb[code * DD + d];
      int addr = ((b * DD + d) << 10) + t0 + r;
      float dv = q - ze[addr];
      sse += dv * dv;
      zq_out[addr] = q;
    }
  }
#pragma unroll
  for (int m = 16; m >= 1; m >>= 1) sse += __shfl_xor(sse, m, 32);
  if (lane == 0) red4[wave] = sse;
  __syncthreads();
  if (tid == 0) {
    float s = red4[0] + red4[1] + red4[2] + red4[3];
    atomicAdd(sseg, s);
  }
}

// ---------------------------------------------------------------------------
// final: perplexity from histogram, loss = 1.5 * SSE / numel
// ---------------------------------------------------------------------------
__global__ void vq_final(const float* __restrict__ counts,
                         const float* __restrict__ sseg,
                         float* __restrict__ out_loss,
                         float* __restrict__ out_ppl) {
  __shared__ float red[1024];
  const int t = threadIdx.x;
  float c = counts[t];
  float e = c / (65536.0f + EPSF);
  red[t] = -e * logf(e + EPSF);
  __syncthreads();
  for (int s = 512; s > 0; s >>= 1) {
    if (t < s) red[t] += red[t + s];
    __syncthreads();
  }
  if (t == 0) {
    *out_ppl  = expf(red[0]);
    *out_loss = 1.5f * (*sseg) / 8388608.0f;
  }
}

// ---------------------------------------------------------------------------
extern "C" void kernel_launch(void* const* d_in, const int* in_sizes, int n_in,
                              void* d_out, int out_size, void* d_ws, size_t ws_size,
                              hipStream_t stream) {
  (void)in_sizes; (void)n_in; (void)out_size; (void)ws_size;
  const float* ze = (const float*)d_in[0];   // z_e  [64,128,1024] f32
  const float* cb = (const float*)d_in[1];   // codebook [1024,128] f32
  float* out = (float*)d_out;

  char* ws = (char*)d_ws;
  _Float16* Bc  = (_Float16*)ws;                     // 1024*256*2 = 524288 B
  float*  cnorm = (float*)(ws + 524288);             // 4 KB
  float*  cnts  = (float*)(ws + 524288 + 4096);      // 4 KB
  float*  sse   = (float*)(ws + 524288 + 8192);      // 4 B

  float* zq   = out;                 // 8388608 floats
  float* loss = out + 8388608;
  float* ppl  = out + 8388609;
  float* idxo = out + 8388610;       // 65536 floats (idx as float)

  vq_prep <<<KK, DD, 0, stream>>>(cb, Bc, cnorm, cnts, sse);
  vq_main <<<NN / MT, 128, 0, stream>>>(ze, cb, Bc, cnorm, cnts, sse, zq, idxo);
  vq_final<<<1, 1024, 0, stream>>>(cnts, sse, loss, ppl);
}
